// MultiHeadAttention_53352083751365
// MI455X (gfx1250) — compile-verified, matrix operand order checked by reading, not run on hardware
//
#include <hip/hip_runtime.h>

// ---------------------------------------------------------------------------
// MHA forward for B=2, S=2048, D=1024, H=16, DH=64 on gfx1250 (CDNA5, wave32).
// All matmuls via v_wmma_f32_16x16x32_bf16. Flash-style streaming softmax.
// K-loops are register double-buffered so loads for step k+1 overlap the
// back-to-back WMMAs of step k.
// ---------------------------------------------------------------------------

#define BB 2
#define SS 2048
#define DD 1024
#define HH 16
#define DH 64

typedef __attribute__((ext_vector_type(16))) __bf16 v16bf;
typedef __attribute__((ext_vector_type(8)))  __bf16 v8bf;
typedef __attribute__((ext_vector_type(4)))  __bf16 v4bf;
typedef __attribute__((ext_vector_type(8)))  float  v8f;

// Load a 16x32 bf16 WMMA fragment (A layout; B uses the mirrored layout with
// column-per-lane, same code). Caller passes p = base + (row0 + lane%16)*ld
// + k0 + koff where koff = (lane>=16 ? 8 : 0). Elements 0-7 <- K koff..koff+7,
// elements 8-15 <- K koff+16..koff+23 (ISA 7.12.2, 16-bit A 16x32).
__device__ __forceinline__ v16bf ld_frag(const __bf16* p) {
    v16bf f;
    *(v8bf*)&f       = *(const v8bf*)(p);
    *((v8bf*)&f + 1) = *(const v8bf*)(p + 16);
    return f;
}

__device__ __forceinline__ v8f wmma_bf16(v16bf a, v16bf b, v8f c) {
    return __builtin_amdgcn_wmma_f32_16x16x32_bf16(
        /*neg_a=*/false, a, /*neg_b=*/false, b,
        /*c_mod=*/(short)0, c, /*reuse_a=*/false, /*reuse_b=*/false);
}

// ---------------------------------------------------------------------------
// Kernel 0: fp32 -> bf16 conversion (vectorized x4)
// ---------------------------------------------------------------------------
__global__ __launch_bounds__(256) void cvt_f32_bf16(
    const float* __restrict__ src, __bf16* __restrict__ dst, int n)
{
    int i = (blockIdx.x * 256 + threadIdx.x) * 4;
    if (i + 3 < n) {
        float4 v = *(const float4*)(src + i);
        v4bf o;
        o.x = (__bf16)v.x; o.y = (__bf16)v.y;
        o.z = (__bf16)v.z; o.w = (__bf16)v.w;
        *(v4bf*)(dst + i) = o;
    }
}

// ---------------------------------------------------------------------------
// Pipelined 16(M) x 64(N) x 1024(K) wave-level GEMM core. A,B row-major bf16
// with K contiguous (B holds the N=64 columns as 4 groups of 16 rows at
// stride 16*ldb). Double-buffered in registers.
// ---------------------------------------------------------------------------
__device__ __forceinline__ void gemm_16x64_core(
    const __bf16* __restrict__ pa,   // = A + (m0+l16)*lda + koff
    const __bf16* __restrict__ pb,   // = B + (n0+l16)*ldb + koff
    size_t ldb, v8f acc[4])
{
    v16bf a  = ld_frag(pa);
    v16bf b0 = ld_frag(pb);
    v16bf b1 = ld_frag(pb + 16 * ldb);
    v16bf b2 = ld_frag(pb + 32 * ldb);
    v16bf b3 = ld_frag(pb + 48 * ldb);
    for (int k0 = 32; k0 < DD; k0 += 32) {
        // issue next-step loads before this step's WMMAs
        v16bf an  = ld_frag(pa + k0);
        v16bf bn0 = ld_frag(pb + k0);
        v16bf bn1 = ld_frag(pb + 16 * ldb + k0);
        v16bf bn2 = ld_frag(pb + 32 * ldb + k0);
        v16bf bn3 = ld_frag(pb + 48 * ldb + k0);
        acc[0] = wmma_bf16(a, b0, acc[0]);
        acc[1] = wmma_bf16(a, b1, acc[1]);
        acc[2] = wmma_bf16(a, b2, acc[2]);
        acc[3] = wmma_bf16(a, b3, acc[3]);
        a = an; b0 = bn0; b1 = bn1; b2 = bn2; b3 = bn3;
    }
    acc[0] = wmma_bf16(a, b0, acc[0]);
    acc[1] = wmma_bf16(a, b1, acc[1]);
    acc[2] = wmma_bf16(a, b2, acc[2]);
    acc[3] = wmma_bf16(a, b3, acc[3]);
}

// ---------------------------------------------------------------------------
// Kernel 1: fused QKV projection.  Out = X @ W^T + bias, per-head layouts:
//   Q,K -> [B,H,S,DH]   V -> transposed [B,H,DH,S]
// grid.x: 48 tiles of 64 cols over 3*1024 fused N; grid.y: 32 blocks of 128 rows.
// ---------------------------------------------------------------------------
__global__ __launch_bounds__(256) void qkv_gemm(
    const __bf16* __restrict__ Xq, const __bf16* __restrict__ Xk,
    const __bf16* __restrict__ Xv,
    const __bf16* __restrict__ Wq, const __bf16* __restrict__ Wk,
    const __bf16* __restrict__ Wv,
    const float* __restrict__ bq, const float* __restrict__ bk,
    const float* __restrict__ bv,
    __bf16* __restrict__ Q, __bf16* __restrict__ Kc, __bf16* __restrict__ Vt)
{
    const int lane = threadIdx.x & 31;
    const int wave = threadIdx.x >> 5;
    const int half = lane >> 4, l16 = lane & 15, koff = half * 8;

    const int m0    = blockIdx.y * 128 + wave * 16;
    const int nglob = blockIdx.x * 64;          // fused N in [0, 3072)
    const int proj  = nglob >> 10;              // 0=Q 1=K 2=V (uniform per block)
    const int nb    = nglob & (DD - 1);

    const __bf16* X    = proj == 0 ? Xq : (proj == 1 ? Xk : Xv);
    const __bf16* W    = proj == 0 ? Wq : (proj == 1 ? Wk : Wv);
    const float*  bias = proj == 0 ? bq : (proj == 1 ? bk : bv);

    v8f acc[4] = {};
    gemm_16x64_core(X + (size_t)(m0 + l16) * DD + koff,
                    W + (size_t)(nb + l16) * DD + koff, DD, acc);

#pragma unroll
    for (int t = 0; t < 4; ++t) {
        int   ncol = nb + t * 16 + l16;
        float bb   = bias[ncol];
        int   h = ncol >> 6, dh = ncol & 63;
#pragma unroll
        for (int i = 0; i < 8; ++i) {
            int m = m0 + half * 8 + i;            // C layout: M = half*8 + i
            int bidx = m >> 11, s = m & (SS - 1);
            __bf16 o = (__bf16)(acc[t][i] + bb);
            if (proj == 2) {
                Vt[(((size_t)bidx * HH + h) * DH + dh) * SS + s] = o;
            } else {
                __bf16* Dst = (proj == 0) ? Q : Kc;
                Dst[(((size_t)bidx * HH + h) * SS + s) * DH + dh] = o;
            }
        }
    }
}

// ---------------------------------------------------------------------------
// Kernel 2: flash attention.  grid.y = B*H (32), grid.x = S/128 (16),
// block = 256 threads (8 waves). Each wave: 16 query rows, streams 64 key
// blocks of 32 keys. Online softmax; P re-shaped C->A layout via per-wave LDS.
// K-fragments double-buffered across key blocks; V-fragments loaded before
// the softmax VALU section so both overlap WMMA/exp work.
// ---------------------------------------------------------------------------
__global__ __launch_bounds__(256) void flash_attn(
    const __bf16* __restrict__ Q, const __bf16* __restrict__ Kc,
    const __bf16* __restrict__ Vt, const unsigned char* __restrict__ mask,
    __bf16* __restrict__ Ctx)
{
    __shared__ __align__(16) __bf16 ldsP[8][16 * 32];

    const int lane = threadIdx.x & 31;
    const int wave = threadIdx.x >> 5;
    const int half = lane >> 4, l16 = lane & 15, koff = half * 8;

    const int bh = blockIdx.y;
    const int b = bh >> 4, h = bh & (HH - 1);
    const int q0 = blockIdx.x * 128 + wave * 16;

    const __bf16* Qb = Q  + (size_t)bh * SS * DH;
    const __bf16* Kb = Kc + (size_t)bh * SS * DH;
    const __bf16* Vb = Vt + (size_t)bh * DH * SS;
    const unsigned char* Mb = mask + (size_t)b * SS * SS;

    // Q fragments for the two 32-wide K-steps of DH=64; live all kernel.
    v16bf aq0 = ld_frag(Qb + (size_t)(q0 + l16) * DH + koff);
    v16bf aq1 = ld_frag(Qb + (size_t)(q0 + l16) * DH + 32 + koff);

    v8f acc[4] = {};
    float mrow[8], lrow[8];
#pragma unroll
    for (int i = 0; i < 8; ++i) { mrow[i] = -3.0e38f; lrow[i] = 0.0f; }

    __bf16* P = ldsP[wave];

    // preload K fragments for key block 0
    const __bf16* pk0 = Kb + (size_t)l16 * DH + koff;
    v16bf k0a = ld_frag(pk0);
    v16bf k0b = ld_frag(pk0 + 32);
    v16bf k1a = ld_frag(pk0 + 16 * DH);
    v16bf k1b = ld_frag(pk0 + 16 * DH + 32);

    for (int kb = 0; kb < SS; kb += 32) {
        // ---- score tiles s0 (keys kb..kb+15), s1 (keys kb+16..kb+31) ----
        v8f s0 = {}, s1 = {};
        s0 = wmma_bf16(aq0, k0a, s0);
        s0 = wmma_bf16(aq1, k0b, s0);
        s1 = wmma_bf16(aq0, k1a, s1);
        s1 = wmma_bf16(aq1, k1b, s1);

        // ---- issue next key-block K loads + this block's V loads now ----
        if (kb + 32 < SS) {
            const __bf16* pkn = Kb + (size_t)(kb + 32 + l16) * DH + koff;
            k0a = ld_frag(pkn);
            k0b = ld_frag(pkn + 32);
            k1a = ld_frag(pkn + 16 * DH);
            k1b = ld_frag(pkn + 16 * DH + 32);
        }
        v16bf bv0 = ld_frag(Vb + (size_t)( 0 + l16) * SS + kb + koff);
        v16bf bv1 = ld_frag(Vb + (size_t)(16 + l16) * SS + kb + koff);
        v16bf bv2 = ld_frag(Vb + (size_t)(32 + l16) * SS + kb + koff);
        v16bf bv3 = ld_frag(Vb + (size_t)(48 + l16) * SS + kb + koff);
        if (kb + 32 < SS) {
            // warm WGP$ for next block's V rows (global_prefetch_b8)
            __builtin_prefetch(Vb + (size_t)( 0 + l16) * SS + kb + 32, 0, 1);
            __builtin_prefetch(Vb + (size_t)(32 + l16) * SS + kb + 32, 0, 1);
        }

        // ---- scale + mask + online softmax update (co-executes w/ XDL) ----
#pragma unroll
        for (int i = 0; i < 8; ++i) {
            int qrow = q0 + half * 8 + i;
            const unsigned char* mp = Mb + (size_t)qrow * SS + kb + l16;
            float v0 = s0[i] * 0.125f;           // 1/sqrt(64)
            float v1 = s1[i] * 0.125f;
            if (mp[0])  v0 = -1.0e9f;
            if (mp[16]) v1 = -1.0e9f;
            float mx = fmaxf(v0, v1);
#pragma unroll
            for (int d = 8; d >= 1; d >>= 1)
                mx = fmaxf(mx, __shfl_xor(mx, d, 32));   // stays in 16-lane half
            float mnew = fmaxf(mrow[i], mx);
            float corr = __expf(mrow[i] - mnew);
            float p0 = __expf(v0 - mnew);
            float p1 = __expf(v1 - mnew);
            float rs = p0 + p1;
#pragma unroll
            for (int d = 8; d >= 1; d >>= 1)
                rs += __shfl_xor(rs, d, 32);
            lrow[i] = lrow[i] * corr + rs;
            mrow[i] = mnew;
#pragma unroll
            for (int t = 0; t < 4; ++t) acc[t][i] *= corr;
            int r = half * 8 + i;                 // C-layout row
            P[r * 32 + l16]      = (__bf16)p0;
            P[r * 32 + 16 + l16] = (__bf16)p1;
        }

        // ---- P (LDS, re-read in A layout) @ V-block -> accumulators ----
        // Same-wave DS ops are in-order: write->read needs no barrier.
        v16bf ap = ld_frag(P + l16 * 32 + koff);
        acc[0] = wmma_bf16(ap, bv0, acc[0]);
        acc[1] = wmma_bf16(ap, bv1, acc[1]);
        acc[2] = wmma_bf16(ap, bv2, acc[2]);
        acc[3] = wmma_bf16(ap, bv3, acc[3]);
    }

    // ---- normalize, write ctx as [B,S,D] bf16 ----
#pragma unroll
    for (int i = 0; i < 8; ++i) {
        float inv = 1.0f / lrow[i];
        int qrow = q0 + half * 8 + i;
        size_t base = ((size_t)b * SS + qrow) * DD + h * DH;
#pragma unroll
        for (int t = 0; t < 4; ++t)
            Ctx[base + t * 16 + l16] = (__bf16)(acc[t][i] * inv);
    }
}

// ---------------------------------------------------------------------------
// Kernel 3: output projection  Out(f32) = Ctx(bf16) @ Wo^T + bo
// grid.x = 1024/64 = 16, grid.y = 4096/128 = 32.
// ---------------------------------------------------------------------------
__global__ __launch_bounds__(256) void out_gemm(
    const __bf16* __restrict__ Ctx, const __bf16* __restrict__ Wo,
    const float* __restrict__ bo, float* __restrict__ Out)
{
    const int lane = threadIdx.x & 31;
    const int wave = threadIdx.x >> 5;
    const int half = lane >> 4, l16 = lane & 15, koff = half * 8;

    const int m0 = blockIdx.y * 128 + wave * 16;
    const int n0 = blockIdx.x * 64;

    v8f acc[4] = {};
    gemm_16x64_core(Ctx + (size_t)(m0 + l16) * DD + koff,
                    Wo  + (size_t)(n0 + l16) * DD + koff, DD, acc);

#pragma unroll
    for (int t = 0; t < 4; ++t) {
        int   n  = n0 + t * 16 + l16;
        float bb = bo[n];
#pragma unroll
        for (int i = 0; i < 8; ++i) {
            int m = m0 + half * 8 + i;
            Out[(size_t)m * DD + n] = acc[t][i] + bb;
        }
    }
}

// ---------------------------------------------------------------------------
// Host launcher
// ---------------------------------------------------------------------------
extern "C" void kernel_launch(void* const* d_in, const int* in_sizes, int n_in,
                              void* d_out, int out_size, void* d_ws, size_t ws_size,
                              hipStream_t stream)
{
    const float* q    = (const float*)d_in[0];
    const float* k    = (const float*)d_in[1];
    const float* v    = (const float*)d_in[2];
    const unsigned char* mask = (const unsigned char*)d_in[3];
    const float* Wq = (const float*)d_in[4];
    const float* bq = (const float*)d_in[5];
    const float* Wk = (const float*)d_in[6];
    const float* bk = (const float*)d_in[7];
    const float* Wv = (const float*)d_in[8];
    const float* bv = (const float*)d_in[9];
    const float* Wo = (const float*)d_in[10];
    const float* bo = (const float*)d_in[11];
    float* out = (float*)d_out;

    const size_t NX = (size_t)BB * SS * DD;   // 4,194,304
    const size_t NW = (size_t)DD * DD;        // 1,048,576

    __bf16* ws = (__bf16*)d_ws;
    __bf16* Xq  = ws;              // activations, bf16
    __bf16* Xk  = Xq  + NX;
    __bf16* Xv  = Xk  + NX;
    __bf16* Wqb = Xv  + NX;        // weights, bf16
    __bf16* Wkb = Wqb + NW;
    __bf16* Wvb = Wkb + NW;
    __bf16* Wob = Wvb + NW;
    __bf16* Qh  = Wob + NW;        // [B,H,S,DH]
    __bf16* Kh  = Qh  + NX;        // [B,H,S,DH]
    __bf16* Vt  = Kh  + NX;        // [B,H,DH,S]
    __bf16* Ctx = Vt  + NX;        // [B,S,D]

    // 0) fp32 -> bf16
    const int nbx = (int)(NX / 1024);   // 256 thr * 4 elem
    const int nbw = (int)(NW / 1024);
    cvt_f32_bf16<<<nbx, 256, 0, stream>>>(q,  Xq,  (int)NX);
    cvt_f32_bf16<<<nbx, 256, 0, stream>>>(k,  Xk,  (int)NX);
    cvt_f32_bf16<<<nbx, 256, 0, stream>>>(v,  Xv,  (int)NX);
    cvt_f32_bf16<<<nbw, 256, 0, stream>>>(Wq, Wqb, (int)NW);
    cvt_f32_bf16<<<nbw, 256, 0, stream>>>(Wk, Wkb, (int)NW);
    cvt_f32_bf16<<<nbw, 256, 0, stream>>>(Wv, Wvb, (int)NW);
    cvt_f32_bf16<<<nbw, 256, 0, stream>>>(Wo, Wob, (int)NW);

    // 1) fused QKV projection
    qkv_gemm<<<dim3(48, 32), 256, 0, stream>>>(
        Xq, Xk, Xv, Wqb, Wkb, Wvb, bq, bk, bv, Qh, Kh, Vt);

    // 2) flash attention
    flash_attn<<<dim3(SS / 128, BB * HH), 256, 0, stream>>>(
        Qh, Kh, Vt, mask, Ctx);

    // 3) output projection
    out_gemm<<<dim3(DD / 64, (BB * SS) / 128), 256, 0, stream>>>(
        Ctx, Wob, bo, out);
}
